// Attention_68186900791341
// MI455X (gfx1250) — matrix operand order repaired and checked by reading, not compiled, not run on hardware
//
#include <hip/hip_runtime.h>
#include <hip/hip_bf16.h>

typedef __attribute__((ext_vector_type(8)))  _Float16 v8h;
typedef __attribute__((ext_vector_type(16))) _Float16 v16h;
typedef __attribute__((ext_vector_type(8)))  float    v8f;

#define B_    4096
#define N_    200
#define K_    64
#define H1_   80
#define H2_   40
#define NPAD  208     // 13 * 16 rows
#define DIN   256     // 4*K concat feature
#define H1K   96      // h1 K padded to 3*32 for GEMM2
#define H2PAD 48      // 3*16 cols for GEMM2 output
#define NEGV  (-4294967295.0f)

// ---- LDS layout (bytes) ----
#define OFF_Q     0
#define OFF_KEY   128
#define OFF_W1T   (OFF_KEY + NPAD * K_ * 2)        // key f16 [208][64]
#define OFF_H1    (OFF_W1T + H1_ * DIN * 2)        // W1^T f16 [80][256]
#define OFF_W2T   (OFF_H1 + NPAD * H1K * 2)        // h1 f16 [208][96]
#define OFF_H2    (OFF_W2T + H2PAD * H1K * 2)      // W2^T f16 [48][96]
#define OFF_WO    (OFF_H2 + NPAD * H2PAD * 2)      // h2 f16 [208][48]
#define OFF_B1    (OFF_WO + H2_ * 4)
#define OFF_A1    (OFF_B1 + H1_ * 4)
#define OFF_B2    (OFF_A1 + H1_ * 4)
#define OFF_A2    (OFF_B2 + H2PAD * 4)
#define OFF_SC    (OFF_A2 + H2PAD * 4)             // score / p [256] f32
#define OFF_RED   (OFF_SC + 256 * 4)               // reduction [256] f32
#define OFF_PART  (OFF_RED + 256 * 4)              // partial out [4][64] f32
#define LDS_BYTES (OFF_PART + 4 * 64 * 4)

static __device__ inline v16h cat16(v8h lo, v8h hi) {
    return __builtin_shufflevector(lo, hi, 0, 1, 2, 3, 4, 5, 6, 7,
                                           8, 9, 10, 11, 12, 13, 14, 15);
}

__global__ void __launch_bounds__(256, 1)
din_attention_kernel(const float* __restrict__ query,
                     const float* __restrict__ key,
                     const float* __restrict__ value,
                     const int*   __restrict__ mask,
                     const float* __restrict__ W1, const float* __restrict__ b1,
                     const float* __restrict__ a1, const float* __restrict__ W2,
                     const float* __restrict__ b2, const float* __restrict__ a2,
                     const float* __restrict__ Wo, const float* __restrict__ bo,
                     float* __restrict__ out)
{
    extern __shared__ __align__(16) char smem[];
    _Float16* qL   = (_Float16*)(smem + OFF_Q);
    _Float16* keyL = (_Float16*)(smem + OFF_KEY);
    _Float16* w1tL = (_Float16*)(smem + OFF_W1T);
    _Float16* h1L  = (_Float16*)(smem + OFF_H1);
    _Float16* w2tL = (_Float16*)(smem + OFF_W2T);
    _Float16* h2L  = (_Float16*)(smem + OFF_H2);
    float*    woL  = (float*)(smem + OFF_WO);
    float*    b1L  = (float*)(smem + OFF_B1);
    float*    a1L  = (float*)(smem + OFF_A1);
    float*    b2L  = (float*)(smem + OFF_B2);
    float*    a2L  = (float*)(smem + OFF_A2);
    float*    scL  = (float*)(smem + OFF_SC);
    float*    redL = (float*)(smem + OFF_RED);
    float*    parL = (float*)(smem + OFF_PART);

    const int t    = threadIdx.x;
    const int b    = blockIdx.x;
    const int lane = t & 31;
    const int wave = t >> 5;
    const int col  = lane & 15;       // fragment column / row index (M or N mod 16)
    const int hh   = lane >> 4;       // lane-half selects K sub-range
    const int rofs = hh ? 8 : 0;      // C/D row offset for this lane-half

    const size_t kvbase = (size_t)b * N_ * K_;

    // -------- Phase 0: load + convert to f16 in LDS --------
    if (t < K_) qL[t] = (_Float16)query[b * K_ + t];
    for (int idx = t; idx < NPAD * K_; idx += 256) {
        int r = idx >> 6, c = idx & 63;
        keyL[idx] = (r < N_) ? (_Float16)key[kvbase + r * K_ + c] : (_Float16)0.f;
    }
    for (int idx = t; idx < H1_ * DIN; idx += 256) {
        int n = idx >> 8, k = idx & 255;
        w1tL[idx] = (_Float16)W1[k * H1_ + n];            // transpose: row n = out col
    }
    for (int idx = t; idx < H2PAD * H1K; idx += 256) {
        int n = idx / H1K, k = idx - n * H1K;
        float v = (n < H2_ && k < H1_) ? W2[k * H2_ + n] : 0.f;
        w2tL[idx] = (_Float16)v;
    }
    for (int idx = t; idx < NPAD * 16; idx += 256) {      // zero h1 pad cols 80..95
        int r = idx >> 4, c = H1_ + (idx & 15);
        h1L[r * H1K + c] = (_Float16)0.f;
    }
    if (t < H1_)   { b1L[t] = b1[t]; a1L[t] = a1[t]; }
    if (t < H2PAD) { b2L[t] = (t < H2_) ? b2[t] : 0.f;
                     a2L[t] = (t < H2_) ? a2[t] : 1.f; }
    if (t < H2_)   woL[t] = Wo[t];
    // warm L2/L0 for the value tensor used in the tail phase
    for (int idx = t * 32; idx < N_ * K_; idx += 256 * 32)
        __builtin_prefetch(value + kvbase + idx, 0, 0);
    __syncthreads();

    // -------- Phase 1: GEMM1  [208 x 256] * [256 x 80] -> h1 (f16) --------
    // q chunks covering all k-tiles for this lane-half
    v8h q00 = *(const v8h*)(qL + 0  + hh * 8);   // inner 0,  halves 0..7
    v8h q01 = *(const v8h*)(qL + 16 + hh * 8);   // inner 0,  halves 8..15
    v8h q10 = *(const v8h*)(qL + 32 + hh * 8);   // inner 32, halves 0..7
    v8h q11 = *(const v8h*)(qL + 48 + hh * 8);   // inner 32, halves 8..15

    for (int mt = wave; mt < 13; mt += 8) {
        const int r = mt * 16 + col;
        const _Float16* krow = keyL + r * K_;
        v16h afr[8];
        #pragma unroll
        for (int kt = 0; kt < 8; ++kt) {
            const int inner = (kt & 1) * 32;
            const v8h qlo = (kt & 1) ? q10 : q00;
            const v8h qhi = (kt & 1) ? q11 : q01;
            v8h alo, ahi;
            const int quad = kt >> 1;           // which emb quadrant this k-tile hits
            if (quad == 0) { alo = qlo; ahi = qhi; }
            else {
                v8h klo = *(const v8h*)(krow + inner + hh * 8);
                v8h khi = *(const v8h*)(krow + inner + 16 + hh * 8);
                if (quad == 1)      { alo = klo;       ahi = khi;       }
                else if (quad == 2) { alo = qlo - klo; ahi = qhi - khi; }
                else                { alo = qlo * klo; ahi = qhi * khi; }
            }
            afr[kt] = cat16(alo, ahi);
        }
        for (int nt = 0; nt < 5; ++nt) {
            const int n = nt * 16 + col;
            const _Float16* wrow = w1tL + n * DIN + hh * 8;
            v8f acc = {};
            #pragma unroll
            for (int kt = 0; kt < 8; ++kt) {
                v8h blo = *(const v8h*)(wrow + kt * 32);
                v8h bhi = *(const v8h*)(wrow + kt * 32 + 16);
                acc = __builtin_amdgcn_wmma_f32_16x16x32_f16(
                          false, afr[kt], false, cat16(blo, bhi),
                          (short)0, acc, false, false);
            }
            const float bb = b1L[n], aa = a1L[n];
            #pragma unroll
            for (int i = 0; i < 8; ++i) {
                float v = acc[i] + bb;
                v = v > 0.f ? v : aa * v;
                h1L[(mt * 16 + rofs + i) * H1K + n] = (_Float16)v;
            }
        }
    }
    __syncthreads();

    // -------- Phase 2: GEMM2  [208 x 96] * [96 x 48] -> h2 (f16) --------
    for (int tt = wave; tt < 39; tt += 8) {
        const int mt = tt / 3, nt = tt - mt * 3;
        const int r = mt * 16 + col;
        const int n = nt * 16 + col;
        const _Float16* arow = h1L + r * H1K;
        const _Float16* wrow = w2tL + n * H1K + hh * 8;
        v8f acc = {};
        #pragma unroll
        for (int kt = 0; kt < 3; ++kt) {
            v8h alo = *(const v8h*)(arow + kt * 32 + hh * 8);
            v8h ahi = *(const v8h*)(arow + kt * 32 + 16 + hh * 8);
            v8h blo = *(const v8h*)(wrow + kt * 32);
            v8h bhi = *(const v8h*)(wrow + kt * 32 + 16);
            acc = __builtin_amdgcn_wmma_f32_16x16x32_f16(
                      false, cat16(alo, ahi), false, cat16(blo, bhi),
                      (short)0, acc, false, false);
        }
        const float bb = b2L[n], aa = a2L[n];
        #pragma unroll
        for (int i = 0; i < 8; ++i) {
            float v = acc[i] + bb;
            v = v > 0.f ? v : aa * v;
            h2L[(mt * 16 + rofs + i) * H2PAD + n] = (_Float16)v;
        }
    }
    __syncthreads();

    // -------- Phase 3: score + masked softmax over N --------
    float sc = NEGV;
    if (t < N_ && mask[b * N_ + t] != 0) {
        float s = bo[0];
        const _Float16* hrow = h2L + t * H2PAD;
        #pragma unroll
        for (int j = 0; j < H2_; ++j) s += (float)hrow[j] * woL[j];
        sc = s;
    }
    scL[t]  = sc;
    redL[t] = sc;
    __syncthreads();
    for (int s = 128; s > 0; s >>= 1) {
        if (t < s) redL[t] = fmaxf(redL[t], redL[t + s]);
        __syncthreads();
    }
    const float mx = redL[0];
    __syncthreads();
    const float e = (t < N_) ? __expf(scL[t] - mx) : 0.f;
    scL[t]  = e;          // unnormalized p
    redL[t] = e;
    __syncthreads();
    for (int s = 128; s > 0; s >>= 1) {
        if (t < s) redL[t] += redL[t + s];
        __syncthreads();
    }
    const float inv = 1.0f / redL[0];
    __syncthreads();

    // -------- Phase 4: out[b,k] = (sum_n p[n] * value[b,n,k]) * inv --------
    const int kk = t & 63;
    const int g  = t >> 6;
    float acc = 0.f;
    const float* vcol = value + kvbase + kk;
    for (int n = g; n < N_; n += 4) acc += scL[n] * vcol[n * K_];
    parL[g * 64 + kk] = acc;
    __syncthreads();
    if (t < 64)
        out[b * K_ + t] =
            (parL[t] + parL[64 + t] + parL[128 + t] + parL[192 + t]) * inv;
}

extern "C" void kernel_launch(void* const* d_in, const int* in_sizes, int n_in,
                              void* d_out, int out_size, void* d_ws, size_t ws_size,
                              hipStream_t stream) {
    (void)in_sizes; (void)n_in; (void)out_size; (void)d_ws; (void)ws_size;
    const float* query = (const float*)d_in[0];
    const float* key   = (const float*)d_in[1];
    const float* value = (const float*)d_in[2];
    const int*   mask  = (const int*)  d_in[3];
    const float* W1    = (const float*)d_in[4];
    const float* b1    = (const float*)d_in[5];
    const float* a1    = (const float*)d_in[6];
    const float* W2    = (const float*)d_in[7];
    const float* b2    = (const float*)d_in[8];
    const float* a2    = (const float*)d_in[9];
    const float* Wo    = (const float*)d_in[10];
    const float* bo    = (const float*)d_in[11];
    float* out = (float*)d_out;

    hipFuncSetAttribute((const void*)din_attention_kernel,
                        hipFuncAttributeMaxDynamicSharedMemorySize, LDS_BYTES);
    din_attention_kernel<<<B_, 256, LDS_BYTES, stream>>>(
        query, key, value, mask, W1, b1, a1, W2, b2, a2, Wo, bo, out);
}
